// NeuromorphicLM_67199058313497
// MI455X (gfx1250) — matrix operand order repaired and checked by reading, not compiled, not run on hardware
//
#include <hip/hip_runtime.h>
#include <cstdint>
#include <cstddef>

// ---------------- CDNA5 (gfx1250) types ----------------
typedef __bf16 bf16_t;
typedef __attribute__((ext_vector_type(16))) __bf16 v16bf;
typedef __attribute__((ext_vector_type(8)))  __bf16 v8bf;
typedef __attribute__((ext_vector_type(4)))  __bf16 v4bf;
typedef __attribute__((ext_vector_type(8)))  float  v8f;

union V16 { v16bf v; v8bf h8[2]; };

// ---- TDM availability probe (device pass only) ----
#if defined(__HIP_DEVICE_COMPILE__) && defined(__has_builtin)
#if __has_builtin(__builtin_amdgcn_tensor_load_to_lds) && \
    __has_builtin(__builtin_amdgcn_s_wait_tensorcnt)
#define HAVE_TDM 1
#else
#define HAVE_TDM 0
#endif
#else
#define HAVE_TDM 0
#endif

__device__ __forceinline__ bf16_t f2bf(float f) {
  unsigned int u = __builtin_bit_cast(unsigned int, f);
  u += 0x7FFFu + ((u >> 16) & 1u);                 // round-to-nearest-even
  unsigned short h = (unsigned short)(u >> 16);
  return __builtin_bit_cast(bf16_t, h);
}

__device__ __forceinline__ v8f wmma_acc(v16bf a, v16bf b, v8f acc) {
  return __builtin_amdgcn_wmma_f32_16x16x32_bf16(false, a, false, b,
                                                 (short)0, acc, false, false);
}

#if HAVE_TDM
typedef __attribute__((ext_vector_type(4))) unsigned int v4u;
typedef __attribute__((ext_vector_type(8))) int v8i;
typedef __attribute__((ext_vector_type(4))) int v4i;

// Issue one TDM 2-D tile load: tile_w x tile_h fp32 elements, global row
// stride row_stride elements, into LDS byte offset lds_off (rows packed
// contiguously). Groups 2/3 zero => <=2D tensor. Tracked by TENSORcnt.
__device__ __forceinline__ void tdm_load_2d(unsigned lds_off, const void* gptr,
                                            unsigned tile_w, unsigned tile_h,
                                            unsigned row_stride) {
  const unsigned long long ga = (unsigned long long)gptr;
  v4u g0;
  g0[0] = 1u;                                                 // count = 1 valid D#
  g0[1] = lds_off;                                            // lds_addr (bytes)
  g0[2] = (unsigned)(ga & 0xffffffffu);                       // global_addr[31:0]
  g0[3] = (unsigned)((ga >> 32) & 0x01ffffffu) | (2u << 30);  // [56:32] | type=2
  v8i g1;
  g1[0] = (int)(2u << 16);                                    // data_size = 4 bytes
  g1[1] = (int)(row_stride << 16);                            // tensor_dim0[15:0]
  g1[2] = (int)((row_stride >> 16) | (16384u << 16));         // dim0[31:16] | dim1[15:0]
  g1[3] = (int)(tile_w << 16);                                // dim1[31:16]=0 | tile_dim0
  g1[4] = (int)(tile_h & 0xffffu);                            // tile_dim1 | tile_dim2=0
  g1[5] = (int)row_stride;                                    // tensor_dim0_stride[31:0]
  g1[6] = 0;
  g1[7] = 0;
  v4i gz = {0, 0, 0, 0};
#if __clang_major__ >= 23
  v8i g4 = {0, 0, 0, 0, 0, 0, 0, 0};
  __builtin_amdgcn_tensor_load_to_lds(g0, g1, gz, gz, g4, 0);
#else
  __builtin_amdgcn_tensor_load_to_lds(g0, g1, gz, gz, 0);
#endif
}
#endif  // HAVE_TDM

// ---------------- model dims ----------------
#define BS   4
#define NSEQ 1024
#define VOC  32000
#define DIM  1024
#define NB   8
#define CC   4
#define DC   64
#define HH   256
#define TT   2048     // NB*CC*DC
#define RPASS 3

// =====================================================================
// 1) fp32 -> bf16 cast (grid stride, float4 in / v4bf out)
// =====================================================================
__global__ void k_cast_bf16(const float4* __restrict__ in, bf16_t* __restrict__ out, int n4) {
  for (int i = blockIdx.x * blockDim.x + threadIdx.x; i < n4; i += gridDim.x * blockDim.x) {
    float4 v = in[i];
    v4bf o; o[0] = f2bf(v.x); o[1] = f2bf(v.y); o[2] = f2bf(v.z); o[3] = f2bf(v.w);
    *(v4bf*)(out + (size_t)i * 4) = o;
  }
}

// =====================================================================
// 2) batched transpose+cast: in fp32 [bt][R][C] -> out bf16 [bt][C][R]
// =====================================================================
__global__ void k_transpose_cast(const float* __restrict__ in, bf16_t* __restrict__ out,
                                 int Rr, int Cc) {
  __shared__ float tile[32][33];
  const int bt = blockIdx.z;
  const float* src = in + (size_t)bt * Rr * Cc;
  bf16_t* dst = out + (size_t)bt * Rr * Cc;
  const int c0 = blockIdx.x * 32, r0 = blockIdx.y * 32;
  for (int i = threadIdx.y; i < 32; i += 8)
    tile[i][threadIdx.x] = src[(size_t)(r0 + i) * Cc + (c0 + threadIdx.x)];
  __syncthreads();
  for (int i = threadIdx.y; i < 32; i += 8)
    dst[(size_t)(c0 + i) * Rr + (r0 + threadIdx.x)] = f2bf(tile[threadIdx.x][i]);
}

// =====================================================================
// 3) embedding gather + positional add -> bf16 activations [BS*N][D]
// =====================================================================
__global__ void __launch_bounds__(256) k_embed(const int* __restrict__ ids,
                                               const float* __restrict__ emb,
                                               const float* __restrict__ pos,
                                               bf16_t* __restrict__ xbf) {
  const int row = blockIdx.x;          // 0..4095
  const int n   = row & (NSEQ - 1);
  const int id  = ids[row];
  const int d   = threadIdx.x * 4;
  float4 e = *(const float4*)(emb + (size_t)id * DIM + d);
  float4 p = *(const float4*)(pos + (size_t)n * DIM + d);
  v4bf o; o[0] = f2bf(e.x + p.x); o[1] = f2bf(e.y + p.y);
          o[2] = f2bf(e.z + p.z); o[3] = f2bf(e.w + p.w);
  *(v4bf*)(xbf + (size_t)row * DIM + d) = o;
}

// =====================================================================
// 4) WMMA GEMM, software-pipelined with peeled epilogue:
//    C[M][N] fp32 = A[M][K]bf16 * Bt[N][K]bf16^T (+bias).
//    Workgroup = 8 waves covering 32(M) x 256(N); wave owns a 16x64 strip.
//    Loop body loads fragments for step kk while multiplying step kk-32,
//    so s_wait_loadcnt is hidden behind WMMA; no clamp/select chains.
// =====================================================================
__global__ void __launch_bounds__(256) k_gemm_bf16(const bf16_t* __restrict__ A,
                                                   const bf16_t* __restrict__ Bt,
                                                   const float* __restrict__ bias,
                                                   float* __restrict__ C,
                                                   int M, int N, int K) {
  (void)M;
  const int lane = threadIdx.x & 31;
  const int wid  = threadIdx.x >> 5;
  const int m16  = lane & 15;
  const int hi   = lane >> 4;
  const int tileM  = (blockIdx.y * 2 + (wid & 1)) * 16;
  const int tileN0 = blockIdx.x * 256 + (wid >> 1) * 64;

  const bf16_t* arow = A + (size_t)(tileM + m16) * K;
  const bf16_t* b0 = Bt + (size_t)(tileN0 +  0 + m16) * K + hi * 16;
  const bf16_t* b1 = Bt + (size_t)(tileN0 + 16 + m16) * K + hi * 16;
  const bf16_t* b2 = Bt + (size_t)(tileN0 + 32 + m16) * K + hi * 16;
  const bf16_t* b3 = Bt + (size_t)(tileN0 + 48 + m16) * K + hi * 16;

  v8f acc0 = {}, acc1 = {}, acc2 = {}, acc3 = {};

  // prologue: fragments for kk = 0
  V16 a, b0v, b1v, b2v, b3v;
  a.h8[0] = *(const v8bf*)(arow + hi * 16);       // contiguous K layout: k = 16*hi + s
  a.h8[1] = *(const v8bf*)(arow + hi * 16 + 8);
  b0v.v = *(const v16bf*)(b0);
  b1v.v = *(const v16bf*)(b1);
  b2v.v = *(const v16bf*)(b2);
  b3v.v = *(const v16bf*)(b3);

  for (int kk = 32; kk < K; kk += 32) {
    __builtin_prefetch(arow + kk + 256, 0, 1);       // global_prefetch_b8
    V16 an, bn0, bn1, bn2, bn3;
    an.h8[0] = *(const v8bf*)(arow + kk + hi * 16);
    an.h8[1] = *(const v8bf*)(arow + kk + hi * 16 + 8);
    bn0.v = *(const v16bf*)(b0 + kk);
    bn1.v = *(const v16bf*)(b1 + kk);
    bn2.v = *(const v16bf*)(b2 + kk);
    bn3.v = *(const v16bf*)(b3 + kk);

    // consume previous fragments (already resident -> no wait before WMMA)
    acc0 = wmma_acc(a.v, b0v.v, acc0);
    acc1 = wmma_acc(a.v, b1v.v, acc1);
    acc2 = wmma_acc(a.v, b2v.v, acc2);
    acc3 = wmma_acc(a.v, b3v.v, acc3);

    a = an; b0v = bn0; b1v = bn1; b2v = bn2; b3v = bn3;
  }

  // epilogue: multiply the final fragments
  acc0 = wmma_acc(a.v, b0v.v, acc0);
  acc1 = wmma_acc(a.v, b1v.v, acc1);
  acc2 = wmma_acc(a.v, b2v.v, acc2);
  acc3 = wmma_acc(a.v, b3v.v, acc3);

  // C/D layout: VGPR j holds row (j + hi*8), column = tileN + lane%16
  #pragma unroll
  for (int t = 0; t < 4; ++t) {
    const v8f acc = (t == 0) ? acc0 : (t == 1) ? acc1 : (t == 2) ? acc2 : acc3;
    const int col = tileN0 + t * 16 + m16;
    const float bv = bias ? bias[col] : 0.f;
    #pragma unroll
    for (int j = 0; j < 8; ++j) {
      const int row = tileM + j + hi * 8;
      C[(size_t)row * N + col] = acc[j] + bv;
    }
  }
}

// =====================================================================
// 5) fused refinement: workgroup = 4 tokens, wave t handles block t.
//    x tile is DMA'd HBM->LDS by the Tensor Data Mover (TENSORcnt), then
//    all 3 passes run out of LDS. hs (bf16 activations) also in LDS.
//    LDS: xs 32768 | hs 65536 | zs 8192 | psum 32 = 106528 B (WGP: 320KB)
// =====================================================================
__global__ void __launch_bounds__(256) k_refine(const float* __restrict__ Xb,
                                                const bf16_t* __restrict__ W1T,  // [8][256][64]
                                                const bf16_t* __restrict__ W2T,  // [8][64][256]
                                                const float* __restrict__ Wz,    // [8][64][64]
                                                const float* __restrict__ lam_logit,
                                                bf16_t* __restrict__ Xb_bf,
                                                float* __restrict__ pcm_part) {
  extern __shared__ char smem[];
  const int lane = threadIdx.x & 31;
  const int wid  = threadIdx.x >> 5;   // block index t
  const int m16  = lane & 15;
  const int hi   = lane >> 4;

  float*  xs = (float*)smem + wid * (16 * DC);                      // [16][64] rows = (tok,c)
  bf16_t* hs = (bf16_t*)(smem + 32768) + wid * (16 * HH);           // [16][256]
  float*  zs = (float*)(smem + 32768 + 65536) + wid * (4 * DC);     // [4][64]
  float*  psum = (float*)(smem + 32768 + 65536 + 8192);             // [8]

  const size_t tok0 = (size_t)blockIdx.x * 4;
  const float lam = 1.f / (1.f + expf(-lam_logit[0]));

  // ---- stage this block's 4-token slab into LDS ----
#if HAVE_TDM
  // TDM 2-D tile: 4 rows x 256 fp32, row stride 2048 elements, packed into
  // xs (dynamic LDS starts at offset 0; this wave's region at wid*4096 B).
  tdm_load_2d((unsigned)(wid * 16 * DC * 4),
              Xb + tok0 * TT + (size_t)wid * (CC * DC),
              /*tile_w=*/CC * DC, /*tile_h=*/4, /*row_stride=*/TT);
  __builtin_amdgcn_s_wait_tensorcnt(0);
#else
  #pragma unroll
  for (int tk = 0; tk < 4; ++tk) {
    const float4* src = (const float4*)(Xb + (tok0 + tk) * TT + wid * (CC * DC));
    float4* dst = (float4*)(xs + tk * 256);
    for (int i = lane; i < 64; i += 32) dst[i] = src[i];
  }
#endif

  const bf16_t* W1b = W1T + (size_t)wid * HH * DC;
  const bf16_t* W2b = W2T + (size_t)wid * DC * HH;
  const float*  Wzb = Wz  + (size_t)wid * DC * DC;

  float zhat_prev[8];
  float local_pcm = 0.f;

  for (int r = 0; r < RPASS; ++r) {
    // ---- z = mean_c(x) (pre-update), pcm against previous z_hat ----
    #pragma unroll
    for (int i = 0; i < 8; ++i) {
      const int e = lane + 32 * i;       // 0..255 -> (tok, d)
      const int tk = e >> 6, d = e & 63;
      const float z = 0.25f * (xs[(tk * 4 + 0) * DC + d] + xs[(tk * 4 + 1) * DC + d] +
                               xs[(tk * 4 + 2) * DC + d] + xs[(tk * 4 + 3) * DC + d]);
      zs[tk * DC + d] = z;
      if (r > 0) { const float df = zhat_prev[i] - z; local_pcm += df * df; }
    }
    // ---- z_hat = z @ Wz[t] (needed for next pass only) ----
    if (r < RPASS - 1) {
      #pragma unroll
      for (int i = 0; i < 8; ++i) {
        const int e = lane + 32 * i;
        const int tk = e >> 6, d = e & 63;
        const float* w = Wzb + d;
        const float* zrow = zs + tk * DC;
        float acc = 0.f;
        for (int dd = 0; dd < DC; ++dd) acc += zrow[dd] * w[dd * DC];
        zhat_prev[i] = acc;
      }
    }
    // ---- h = gelu(x @ W1[t]) : [16x64]@[64x256] via WMMA ----
    for (int nt = 0; nt < 16; ++nt) {
      v8f acc = {};
      #pragma unroll
      for (int ks = 0; ks < 2; ++ks) {
        const int kk = ks * 32;
        V16 a;
        #pragma unroll
        for (int j = 0; j < 8; ++j) a.v[j]     = f2bf(xs[m16 * DC + kk + hi * 16 + j]);
        #pragma unroll
        for (int j = 0; j < 8; ++j) a.v[8 + j] = f2bf(xs[m16 * DC + kk + hi * 16 + 8 + j]);
        V16 b; b.v = *(const v16bf*)(W1b + (size_t)(nt * 16 + m16) * DC + kk + hi * 16);
        acc = wmma_acc(a.v, b.v, acc);
      }
      const int n = nt * 16 + m16;
      #pragma unroll
      for (int j = 0; j < 8; ++j) {
        const int rrow = j + hi * 8;
        const float x = acc[j];
        const float g = 0.5f * x * (1.f + tanhf(0.7978845608f * (x + 0.044715f * x * x * x)));
        hs[rrow * HH + n] = f2bf(g);
      }
    }
    // ---- x += coeff * (h @ W2[t]) : [16x256]@[256x64] via WMMA ----
    const float coeff = (r == 0) ? 1.f : lam;   // blend = x + lam*delta
    for (int nt = 0; nt < 4; ++nt) {
      v8f acc = {};
      #pragma unroll
      for (int ks = 0; ks < 8; ++ks) {
        const int kk = ks * 32;
        V16 a;
        a.h8[0] = *(const v8bf*)(hs + m16 * HH + kk + hi * 16);
        a.h8[1] = *(const v8bf*)(hs + m16 * HH + kk + hi * 16 + 8);
        V16 b; b.v = *(const v16bf*)(W2b + (size_t)(nt * 16 + m16) * HH + kk + hi * 16);
        acc = wmma_acc(a.v, b.v, acc);
      }
      const int dcol = nt * 16 + m16;
      #pragma unroll
      for (int j = 0; j < 8; ++j) xs[(j + hi * 8) * DC + dcol] += coeff * acc[j];
    }
  }

  // ---- write refined tile back as bf16 (feeds fan-in GEMM) ----
  #pragma unroll
  for (int tk = 0; tk < 4; ++tk) {
    bf16_t* dst = Xb_bf + (tok0 + tk) * TT + wid * (CC * DC);
    const float4* src = (const float4*)(xs + tk * 256);
    for (int i = lane; i < 64; i += 32) {
      float4 v = src[i];
      v4bf o; o[0] = f2bf(v.x); o[1] = f2bf(v.y); o[2] = f2bf(v.z); o[3] = f2bf(v.w);
      *(v4bf*)(dst + i * 4) = o;
    }
  }

  // ---- pcm partial reduction (wave shuffle + LDS) ----
  #pragma unroll
  for (int o = 16; o > 0; o >>= 1) local_pcm += __shfl_xor(local_pcm, o, 32);
  if (lane == 0) psum[wid] = local_pcm;
  __syncthreads();
  if (threadIdx.x == 0) {
    float s = 0.f;
    #pragma unroll
    for (int i = 0; i < 8; ++i) s += psum[i];
    pcm_part[blockIdx.x] = s;
  }
}

// =====================================================================
// 6) LayerNorm over D=1024, one row per workgroup -> bf16
// =====================================================================
__global__ void __launch_bounds__(256) k_layernorm(const float* __restrict__ X,
                                                   const float* __restrict__ lng,
                                                   const float* __restrict__ lnb,
                                                   bf16_t* __restrict__ out) {
  __shared__ float red[16];
  const int row = blockIdx.x, tid = threadIdx.x;
  float4 v = *((const float4*)(X + (size_t)row * DIM) + tid);
  float s = v.x + v.y + v.z + v.w;
  float q = v.x * v.x + v.y * v.y + v.z * v.z + v.w * v.w;
  #pragma unroll
  for (int o = 16; o > 0; o >>= 1) { s += __shfl_xor(s, o, 32); q += __shfl_xor(q, o, 32); }
  if ((tid & 31) == 0) { red[tid >> 5] = s; red[8 + (tid >> 5)] = q; }
  __syncthreads();
  float ts = 0.f, tq = 0.f;
  #pragma unroll
  for (int i = 0; i < 8; ++i) { ts += red[i]; tq += red[8 + i]; }
  const float mu = ts * (1.f / DIM);
  const float var = tq * (1.f / DIM) - mu * mu;
  const float rs = rsqrtf(var + 1e-5f);
  const int d = tid * 4;
  v4bf o;
  o[0] = f2bf((v.x - mu) * rs * lng[d + 0] + lnb[d + 0]);
  o[1] = f2bf((v.y - mu) * rs * lng[d + 1] + lnb[d + 1]);
  o[2] = f2bf((v.z - mu) * rs * lng[d + 2] + lnb[d + 2]);
  o[3] = f2bf((v.w - mu) * rs * lng[d + 3] + lnb[d + 3]);
  *(v4bf*)(out + (size_t)row * DIM + d) = o;
}

// =====================================================================
// 7) aux finalize: sum 1024 partials, scale by PCM_W / (BS*N*Dc)
// =====================================================================
__global__ void k_finalize(const float* __restrict__ part, float* __restrict__ aux) {
  __shared__ float red[8];
  const int tid = threadIdx.x;
  float s = part[tid] + part[tid + 256] + part[tid + 512] + part[tid + 768];
  #pragma unroll
  for (int o = 16; o > 0; o >>= 1) s += __shfl_xor(s, o, 32);
  if ((tid & 31) == 0) red[tid >> 5] = s;
  __syncthreads();
  if (tid == 0) {
    float t = 0.f;
    #pragma unroll
    for (int i = 0; i < 8; ++i) t += red[i];
    aux[0] = t * (0.1f / (float)(BS * NSEQ * DC));
  }
}

// =====================================================================
// launch
// =====================================================================
extern "C" void kernel_launch(void* const* d_in, const int* in_sizes, int n_in,
                              void* d_out, int out_size, void* d_ws, size_t ws_size,
                              hipStream_t stream) {
  (void)in_sizes; (void)n_in; (void)ws_size;
  const int*   ids   = (const int*)  d_in[0];
  const float* emb   = (const float*)d_in[1];
  const float* pos   = (const float*)d_in[2];
  const float* fo_w  = (const float*)d_in[3];
  const float* fo_b  = (const float*)d_in[4];
  const float* W1    = (const float*)d_in[5];
  const float* W2    = (const float*)d_in[6];
  const float* Wz    = (const float*)d_in[7];
  const float* fi_w  = (const float*)d_in[8];
  const float* fi_b  = (const float*)d_in[9];
  const float* ln_g  = (const float*)d_in[10];
  const float* ln_b  = (const float*)d_in[11];
  const float* lamlg = (const float*)d_in[12];

  float* logits = (float*)d_out;                 // [4096][32000]
  float* aux    = logits + (size_t)out_size - 1; // trailing scalar

  // ---- carve workspace (256B aligned) ----
  size_t off = 0;
  auto carve = [&](size_t bytes) {
    void* p = (char*)d_ws + off;
    off += (bytes + 255) & ~(size_t)255;
    return p;
  };
  bf16_t* emb_bf = (bf16_t*)carve((size_t)VOC * DIM * 2);   // [V][D] = lm-head B[N][K]
  bf16_t* x_bf   = (bf16_t*)carve((size_t)4096 * DIM * 2);  // embed output
  bf16_t* fo_wT  = (bf16_t*)carve((size_t)TT * DIM * 2);    // [T][D]
  bf16_t* fi_wT  = (bf16_t*)carve((size_t)DIM * TT * 2);    // [D][T]
  bf16_t* W1T    = (bf16_t*)carve((size_t)NB * HH * DC * 2);// [8][256][64]
  bf16_t* W2T    = (bf16_t*)carve((size_t)NB * DC * HH * 2);// [8][64][256]
  float*  Xb     = (float*) carve((size_t)4096 * TT * 4);   // fan-out output fp32
  bf16_t* Xb_bf  = (bf16_t*)carve((size_t)4096 * TT * 2);   // refined, bf16
  float*  Xf     = (float*) carve((size_t)4096 * DIM * 4);  // fan-in output fp32
  bf16_t* xln_bf = (bf16_t*)carve((size_t)4096 * DIM * 2);  // post-LN bf16
  float*  pcm    = (float*) carve((size_t)1024 * 4);        // pcm partials

  dim3 tb(32, 8);

  // weight preprocessing (cheap, fully parallel)
  k_cast_bf16<<<8192, 256, 0, stream>>>((const float4*)emb, emb_bf, VOC * DIM / 4);
  k_transpose_cast<<<dim3(TT / 32, DIM / 32, 1), tb, 0, stream>>>(fo_w, fo_wT, DIM, TT);
  k_transpose_cast<<<dim3(DIM / 32, TT / 32, 1), tb, 0, stream>>>(fi_w, fi_wT, TT, DIM);
  k_transpose_cast<<<dim3(HH / 32, DC / 32, NB), tb, 0, stream>>>(W1, W1T, DC, HH);
  k_transpose_cast<<<dim3(DC / 32, HH / 32, NB), tb, 0, stream>>>(W2, W2T, HH, DC);

  // embed + pos
  k_embed<<<4096, 256, 0, stream>>>(ids, emb, pos, x_bf);

  // fan-out: [4096x1024] @ [1024x2048] + fo_b
  k_gemm_bf16<<<dim3(TT / 256, 4096 / 32), 256, 0, stream>>>(
      x_bf, fo_wT, fo_b, Xb, 4096, TT, DIM);

  // fused 3-pass refinement (+ pcm partials, + bf16 output for fan-in)
  const size_t refine_lds = 32768 + 65536 + 8192 + 32;
  k_refine<<<1024, 256, refine_lds, stream>>>(Xb, W1T, W2T, Wz, lamlg, Xb_bf, pcm);

  // fan-in: [4096x2048] @ [2048x1024] + fi_b
  k_gemm_bf16<<<dim3(DIM / 256, 4096 / 32), 256, 0, stream>>>(
      Xb_bf, fi_wT, fi_b, Xf, 4096, DIM, TT);

  // final LayerNorm -> bf16
  k_layernorm<<<4096, 256, 0, stream>>>(Xf, ln_g, ln_b, xln_bf);

  // tied lm-head: [4096x1024] @ [1024x32000] -> fp32 logits (store-bound)
  k_gemm_bf16<<<dim3(VOC / 256, 4096 / 32), 256, 0, stream>>>(
      xln_bf, emb_bf, nullptr, logits, 4096, VOC, DIM);

  // aux scalar
  k_finalize<<<1, 256, 0, stream>>>(pcm, aux);
}
